// _SequenceMixerBlock_61203283968399
// MI455X (gfx1250) — matrix operand order repaired and checked by reading, not compiled, not run on hardware
//
#include <hip/hip_runtime.h>
#include <math.h>

typedef __bf16 bf16_t;
typedef __attribute__((ext_vector_type(16))) __bf16 v16bf;
typedef __attribute__((ext_vector_type(8)))  __bf16 v8bf;
typedef __attribute__((ext_vector_type(8)))  float  v8f;

#define BLK_M 128
#define BLK_N 128
#define BK    32
#define LDSS  40   // 32 + 8 pad; 80B row stride keeps 16B alignment, breaks bank conflicts

enum { EPI_SIGMOID = 0, EPI_TANH = 1, EPI_GELU = 2, EPI_RES = 3 };

__device__ __forceinline__ bf16_t f2bf(float f) { return (bf16_t)f; }

// low 32 bits of a flat LDS pointer == LDS byte address (ISA 10.2 aperture rule)
__device__ __forceinline__ unsigned lds_addr(const void* p) {
  return (unsigned)(unsigned long long)p;
}
__device__ __forceinline__ void async_copy_b128(unsigned lds, const void* gptr) {
  asm volatile("global_load_async_to_lds_b128 %0, %1, off"
               :: "v"(lds), "v"((unsigned long long)gptr) : "memory");
}
__device__ __forceinline__ void wait_async0() {
  asm volatile("s_wait_asynccnt 0x0" ::: "memory");
}

// ---------------------------------------------------------------------------
// One-shot: W[K,N] f32 -> Wt[N,K] bf16 (tiled LDS transpose, both sides coalesced)
// ---------------------------------------------------------------------------
__global__ __launch_bounds__(256) void convert_transpose_kernel(
    const float* __restrict__ W, bf16_t* __restrict__ Wt, int K, int N)
{
  __shared__ bf16_t tile[64][66];               // +2 pad: conflict-free transpose
  const int kb = blockIdx.x * 64, nb = blockIdx.y * 64;
  const int c = threadIdx.x & 63, r4 = threadIdx.x >> 6;  // 4 rows per pass
#pragma unroll
  for (int i = 0; i < 64; i += 4) {
    int k = i + r4;
    tile[k][c] = f2bf(W[(size_t)(kb + k) * N + nb + c]);
  }
  __syncthreads();
#pragma unroll
  for (int i = 0; i < 64; i += 4) {
    int n = i + r4;
    Wt[(size_t)(nb + n) * K + kb + c] = tile[c][n];
  }
}

// ---------------------------------------------------------------------------
// LayerNorm (f32 in) -> bf16 out, one row (D elems) per 256-thread block
// ---------------------------------------------------------------------------
__global__ __launch_bounds__(256) void ln_to_bf16_kernel(
    const float* __restrict__ x, const float* __restrict__ g,
    const float* __restrict__ b, bf16_t* __restrict__ h, int D)
{
  const int row  = blockIdx.x;
  const int tid  = threadIdx.x;
  const int lane = tid & 31;
  const int wave = tid >> 5;
  const int NIT  = D >> 8;

  float vals[8];
  float s = 0.f, s2 = 0.f;
  for (int i = 0; i < NIT; ++i) {
    float v = x[(size_t)row * D + i * 256 + tid];
    vals[i] = v; s += v; s2 += v * v;
  }
  for (int o = 16; o > 0; o >>= 1) {
    s  += __shfl_xor(s,  o, 32);
    s2 += __shfl_xor(s2, o, 32);
  }
  __shared__ float red[18];
  if (lane == 0) { red[wave] = s; red[8 + wave] = s2; }
  __syncthreads();
  if (tid == 0) {
    float ts = 0.f, ts2 = 0.f;
    for (int k = 0; k < 8; ++k) { ts += red[k]; ts2 += red[8 + k]; }
    float mu  = ts / (float)D;
    float var = ts2 / (float)D - mu * mu;
    red[16] = mu;
    red[17] = rsqrtf(var + 1e-5f);
  }
  __syncthreads();
  const float mu = red[16], rstd = red[17];
  for (int i = 0; i < NIT; ++i) {
    int col = i * 256 + tid;
    h[(size_t)row * D + col] = f2bf((vals[i] - mu) * rstd * g[col] + b[col]);
  }
}

// ---------------------------------------------------------------------------
// bf16 WMMA GEMM:  Out = epi(A[M,K] * Bt[N,K]^T + bias [+ res])
// 128x128 block tile, BK=32, 8 waves (4x2) each owning 32x64 (2x4 WMMA tiles),
// double-buffered LDS staged with global_load_async_to_lds_b128 (ASYNCcnt).
// ---------------------------------------------------------------------------
template <int EPI>
__global__ __launch_bounds__(256) void gemm_bf16_kernel(
    const bf16_t* __restrict__ A, const bf16_t* __restrict__ Bt,
    const float* __restrict__ bias, const float* __restrict__ residual,
    void* __restrict__ OutP, int M, int K, int N)
{
  __shared__ __align__(16) bf16_t As[2][BLK_M * LDSS];
  __shared__ __align__(16) bf16_t Bs[2][BLK_N * LDSS];

  const int tid   = threadIdx.x;
  const int wave  = tid >> 5;
  const int lane  = tid & 31;
  const int lrow  = lane & 15;
  const int lhalf = lane >> 4;
  const int m0 = blockIdx.x * BLK_M;
  const int n0 = blockIdx.y * BLK_N;
  const int wm = (wave & 3) * 32;      // 4 wave-rows of 32
  const int wn = (wave >> 2) * 64;     // 2 wave-cols of 64

  // staging map: 16B chunks; A and B tiles are both 128 rows x 32 bf16.
  const int sr = tid >> 2;             // 0..63
  const int sc = (tid & 3) * 8;        // 0,8,16,24
  const bf16_t* gA0 = A  + (size_t)(m0 + sr)      * K + sc;
  const bf16_t* gA1 = A  + (size_t)(m0 + sr + 64) * K + sc;
  const bf16_t* gB0 = Bt + (size_t)(n0 + sr)      * K + sc;
  const bf16_t* gB1 = Bt + (size_t)(n0 + sr + 64) * K + sc;

  unsigned lA0[2], lA1[2], lB0[2], lB1[2];
#pragma unroll
  for (int sbuf = 0; sbuf < 2; ++sbuf) {
    lA0[sbuf] = lds_addr(&As[sbuf][sr * LDSS + sc]);
    lA1[sbuf] = lds_addr(&As[sbuf][(sr + 64) * LDSS + sc]);
    lB0[sbuf] = lds_addr(&Bs[sbuf][sr * LDSS + sc]);
    lB1[sbuf] = lds_addr(&Bs[sbuf][(sr + 64) * LDSS + sc]);
  }

  v8f acc[2][4] = {};

  // prologue: stage tile 0
  async_copy_b128(lA0[0], gA0);
  async_copy_b128(lA1[0], gA1);
  async_copy_b128(lB0[0], gB0);
  async_copy_b128(lB1[0], gB1);
  wait_async0();
  __syncthreads();

  int cur = 0;
  for (int k0 = 0; k0 < K; k0 += BK) {
    const int nxt = cur ^ 1;
    if (k0 + BK < K) {                 // stage next tile while computing this one
      async_copy_b128(lA0[nxt], gA0 + k0 + BK);
      async_copy_b128(lA1[nxt], gA1 + k0 + BK);
      async_copy_b128(lB0[nxt], gB0 + k0 + BK);
      async_copy_b128(lB1[nxt], gB1 + k0 + BK);
    }

    // fragments (ISA 16-bit A 16x32 layout)
    v16bf afrag[2], bfrag[4];
#pragma unroll
    for (int i = 0; i < 2; ++i) {
      int r = wm + i * 16 + lrow;
      v8bf lo = *(const v8bf*)&As[cur][r * LDSS + lhalf * 8];
      v8bf hi = *(const v8bf*)&As[cur][r * LDSS + 16 + lhalf * 8];
      afrag[i] = __builtin_shufflevector(lo, hi, 0,1,2,3,4,5,6,7,
                                                 8,9,10,11,12,13,14,15);
    }
#pragma unroll
    for (int j = 0; j < 4; ++j) {
      int c = wn + j * 16 + lrow;
      v8bf lo = *(const v8bf*)&Bs[cur][c * LDSS + lhalf * 8];
      v8bf hi = *(const v8bf*)&Bs[cur][c * LDSS + 16 + lhalf * 8];
      bfrag[j] = __builtin_shufflevector(lo, hi, 0,1,2,3,4,5,6,7,
                                                 8,9,10,11,12,13,14,15);
    }
#pragma unroll
    for (int i = 0; i < 2; ++i)
#pragma unroll
      for (int j = 0; j < 4; ++j)
        acc[i][j] = __builtin_amdgcn_wmma_f32_16x16x32_bf16(
            false, afrag[i], false, bfrag[j], (short)0, acc[i][j], false, false);

    wait_async0();                     // next tile landed in LDS
    __syncthreads();                   // all waves done reading cur / writing nxt
    cur = nxt;
  }

  // epilogue (C layout: VGPR r -> M = lhalf*8 + r, N = lane&15)
  float*  OutF = (float*)OutP;
  bf16_t* OutB = (bf16_t*)OutP;
#pragma unroll
  for (int i = 0; i < 2; ++i)
#pragma unroll
    for (int j = 0; j < 4; ++j) {
      int col = n0 + wn + j * 16 + lrow;
      float bval = bias[col];
#pragma unroll
      for (int r = 0; r < 8; ++r) {
        int row = m0 + wm + i * 16 + lhalf * 8 + r;
        size_t o = (size_t)row * N + col;
        float v = acc[i][j][r] + bval;
        if constexpr (EPI == EPI_SIGMOID) {
          OutF[o] = 1.f / (1.f + __expf(-v));
        } else if constexpr (EPI == EPI_TANH) {
          OutF[o] = tanhf(v);
        } else if constexpr (EPI == EPI_GELU) {
          OutB[o] = f2bf(0.5f * v * (1.f + erff(v * 0.70710678118f)));
        } else {
          OutF[o] = v + residual[o];
        }
      }
    }
}

// ---------------------------------------------------------------------------
// Segmented linear-recurrence scan: s_t = a_t*s + (1-a_t)*b_t ; y = c_t*s_t
// ---------------------------------------------------------------------------
__global__ void scan_pass1_kernel(const float* __restrict__ ga,
                                  const float* __restrict__ gb,
                                  float* __restrict__ Pseg,
                                  float* __restrict__ Qseg,
                                  int BD, int D, int SEGLEN, int SEG)
{
  int gidx = blockIdx.x * blockDim.x + threadIdx.x;
  if (gidx >= BD * SEG) return;
  int s = gidx / BD, ch = gidx % BD;
  int bb = ch / D, dd = ch % D;
  size_t base = ((size_t)bb * SEG * SEGLEN + (size_t)s * SEGLEN) * D + dd;
  float P = 1.f, Q = 0.f;
  for (int t = 0; t < SEGLEN; ++t) {
    size_t idx = base + (size_t)t * D;
    float a = ga[idx], bv = gb[idx];
    P = a * P;
    Q = a * Q + (1.f - a) * bv;
  }
  Pseg[gidx] = P; Qseg[gidx] = Q;
}

__global__ void scan_carry_kernel(float* __restrict__ Pseg,
                                  const float* __restrict__ Qseg,
                                  int BD, int SEG)
{
  int ch = blockIdx.x * blockDim.x + threadIdx.x;
  if (ch >= BD) return;
  float state = 0.f;
  for (int s = 0; s < SEG; ++s) {
    float P = Pseg[s * BD + ch], Q = Qseg[s * BD + ch];
    Pseg[s * BD + ch] = state;
    state = P * state + Q;
  }
}

__global__ void scan_pass3_kernel(const float* __restrict__ x,
                                  const float* __restrict__ ga,
                                  const float* __restrict__ gb,
                                  const float* __restrict__ gc,
                                  const float* __restrict__ Carry,
                                  float* __restrict__ x2,
                                  int BD, int D, int SEGLEN, int SEG)
{
  int gidx = blockIdx.x * blockDim.x + threadIdx.x;
  if (gidx >= BD * SEG) return;
  int s = gidx / BD, ch = gidx % BD;
  int bb = ch / D, dd = ch % D;
  size_t base = ((size_t)bb * SEG * SEGLEN + (size_t)s * SEGLEN) * D + dd;
  float state = Carry[gidx];
  for (int t = 0; t < SEGLEN; ++t) {
    size_t idx = base + (size_t)t * D;
    float a = ga[idx], bv = gb[idx], c = gc[idx];
    state = a * state + (1.f - a) * bv;
    x2[idx] = x[idx] + c * state;
  }
}

// ---------------------------------------------------------------------------
extern "C" void kernel_launch(void* const* d_in, const int* in_sizes, int n_in,
                              void* d_out, int out_size, void* d_ws, size_t ws_size,
                              hipStream_t stream)
{
  const float* x   = (const float*)d_in[0];
  const float* g1  = (const float*)d_in[1];
  const float* b1  = (const float*)d_in[2];
  const float* Wa  = (const float*)d_in[3];
  const float* ba  = (const float*)d_in[4];
  const float* Wb  = (const float*)d_in[5];
  const float* bbv = (const float*)d_in[6];
  const float* Wc  = (const float*)d_in[7];
  const float* bc  = (const float*)d_in[8];
  const float* g2  = (const float*)d_in[9];
  const float* b2  = (const float*)d_in[10];
  const float* W1  = (const float*)d_in[11];
  const float* bm1 = (const float*)d_in[12];
  const float* W2  = (const float*)d_in[13];
  const float* bm2 = (const float*)d_in[14];
  float* out = (float*)d_out;

  const int Bn = 4, S = 2048, D = 1024, H = 4096;
  const int Ntok = Bn * S;
  const int SEG = 16, SEGLEN = S / SEG, BD = Bn * D;

  // workspace layout
  char* ws = (char*)d_ws;
  size_t off = 0;
  bf16_t* hbuf = (bf16_t*)(ws + off); off += (size_t)Ntok * D * sizeof(bf16_t);
  float* gaB = (float*)(ws + off); off += (size_t)Ntok * D * sizeof(float);
  float* gbB = (float*)(ws + off); off += (size_t)Ntok * D * sizeof(float);
  float* gcB = (float*)(ws + off); off += (size_t)Ntok * D * sizeof(float);
  float* Pseg = (float*)(ws + off); off += (size_t)BD * SEG * sizeof(float);
  float* Qseg = (float*)(ws + off); off += (size_t)BD * SEG * sizeof(float);
  bf16_t* WtA = (bf16_t*)(ws + off); off += (size_t)D * D * sizeof(bf16_t);
  bf16_t* WtB = (bf16_t*)(ws + off); off += (size_t)D * D * sizeof(bf16_t);
  bf16_t* WtC = (bf16_t*)(ws + off); off += (size_t)D * D * sizeof(bf16_t);
  bf16_t* Wt1 = (bf16_t*)(ws + off); off += (size_t)D * H * sizeof(bf16_t);
  bf16_t* Wt2 = (bf16_t*)(ws + off); off += (size_t)H * D * sizeof(bf16_t);
  bf16_t* mlp = (bf16_t*)gaB;   // 64 MB, aliases ga+gb (dead after scan)

  // 0) weight conversion to transposed bf16 [N,K]
  convert_transpose_kernel<<<dim3(D / 64, D / 64), 256, 0, stream>>>(Wa, WtA, D, D);
  convert_transpose_kernel<<<dim3(D / 64, D / 64), 256, 0, stream>>>(Wb, WtB, D, D);
  convert_transpose_kernel<<<dim3(D / 64, D / 64), 256, 0, stream>>>(Wc, WtC, D, D);
  convert_transpose_kernel<<<dim3(D / 64, H / 64), 256, 0, stream>>>(W1, Wt1, D, H);
  convert_transpose_kernel<<<dim3(H / 64, D / 64), 256, 0, stream>>>(W2, Wt2, H, D);

  // 1) LN1 -> bf16
  ln_to_bf16_kernel<<<Ntok, 256, 0, stream>>>(x, g1, b1, hbuf, D);

  // 2) gate GEMMs with fused activations
  dim3 gg(Ntok / BLK_M, D / BLK_N);
  gemm_bf16_kernel<EPI_SIGMOID><<<gg, 256, 0, stream>>>(hbuf, WtA, ba,  nullptr, gaB, Ntok, D, D);
  gemm_bf16_kernel<EPI_TANH   ><<<gg, 256, 0, stream>>>(hbuf, WtB, bbv, nullptr, gbB, Ntok, D, D);
  gemm_bf16_kernel<EPI_SIGMOID><<<gg, 256, 0, stream>>>(hbuf, WtC, bc,  nullptr, gcB, Ntok, D, D);

  // 3) segmented scan; x2 = x + c*state written straight into d_out
  scan_pass1_kernel<<<(BD * SEG) / 256, 256, 0, stream>>>(gaB, gbB, Pseg, Qseg, BD, D, SEGLEN, SEG);
  scan_carry_kernel<<<(BD + 255) / 256, 256, 0, stream>>>(Pseg, Qseg, BD, SEG);
  scan_pass3_kernel<<<(BD * SEG) / 256, 256, 0, stream>>>(x, gaB, gbB, gcB, Pseg, out, BD, D, SEGLEN, SEG);

  // 4) LN2 (reads x2 in d_out) -> bf16 (reuse hbuf)
  ln_to_bf16_kernel<<<Ntok, 256, 0, stream>>>(out, g2, b2, hbuf, D);

  // 5) MLP up-proj + exact GELU -> bf16
  dim3 gm1(Ntok / BLK_M, H / BLK_N);
  gemm_bf16_kernel<EPI_GELU><<<gm1, 256, 0, stream>>>(hbuf, Wt1, bm1, nullptr, mlp, Ntok, D, H);

  // 6) MLP down-proj + bias + residual -> d_out (f32)
  dim3 gm2(Ntok / BLK_M, D / BLK_N);
  gemm_bf16_kernel<EPI_RES><<<gm2, 256, 0, stream>>>(mlp, Wt2, bm2, out, out, Ntok, H, D);
}
